// Jeffrey_64072322122033
// MI455X (gfx1250) — matrix operand order repaired and checked by reading, not compiled, not run on hardware
//
#include <hip/hip_runtime.h>
#include <math.h>

// ---------------------------------------------------------------------------
// Jeffrey loss forward on MI455X (gfx1250, wave32, WMMA).
//   inputs:  d_in[0] = X  [4096 x 1024] fp32
//            d_in[1] = t  [4096] int32 (class ids 0..63)
//   output:  d_out[0] scalar fp32 loss
// ---------------------------------------------------------------------------

typedef __attribute__((ext_vector_type(16))) _Float16 v16h;
typedef __attribute__((ext_vector_type(8)))  float    v8f;

constexpr int   N_     = 4096;
constexpr int   D_     = 1024;
constexpr int   NC_    = 64;
constexpr float EPS_   = 0.1f;
constexpr int   TILE   = 128;
constexpr int   KT     = 32;            // K-step (f16 WMMA K)
constexpr int   NT     = N_ / TILE;     // 32 tiles per side
constexpr int   NTILES = NT * (NT + 1) / 2;  // 528 upper-tri tiles
constexpr int   LSTR   = 40;            // LDS row stride (halves): 80B rows,
                                        // 16B aligned, conflict-friendly banks

__device__ __forceinline__ float softplus_f(float x) {
    // stable: max(x,0) + log1p(exp(-|x|))
    return fmaxf(x, 0.f) + log1pf(expf(-fabsf(x)));
}

// ---------------------------------------------------------------------------
// K1: per-class sums S_c[d] and sum-of-squares Q_c[d]; class counts.
// One block per class, deterministic (no float atomics). X read from L2.
// ---------------------------------------------------------------------------
__global__ __launch_bounds__(256) void class_stats_kernel(
    const float* __restrict__ X, const int* __restrict__ tgt,
    float* __restrict__ Sc, float* __restrict__ Qc, int* __restrict__ cnt)
{
    const int c   = blockIdx.x;
    const int tid = threadIdx.x;
    __shared__ int tS[N_];                       // 16 KB
    for (int i = tid; i < N_; i += 256) tS[i] = tgt[i];
    __syncthreads();

    float s[4]  = {0.f, 0.f, 0.f, 0.f};
    float qq[4] = {0.f, 0.f, 0.f, 0.f};
    int cc = 0;
    for (int i = 0; i < N_; ++i) {
        if (tS[i] == c) {                        // uniform branch per block
            ++cc;
            const float* row = X + (size_t)i * D_;
            #pragma unroll
            for (int u = 0; u < 4; ++u) {
                float v = row[tid + u * 256];
                s[u]  += v;
                qq[u] += v * v;
            }
        }
    }
    #pragma unroll
    for (int u = 0; u < 4; ++u) {
        Sc[(size_t)c * D_ + tid + u * 256] = s[u];
        Qc[(size_t)c * D_ + tid + u * 256] = qq[u];
    }
    if (tid == 0) cnt[c] = cc;
}

// ---------------------------------------------------------------------------
// K2: per-dim weights w_d = 1/sigma0 - 1/sigma1.
// ---------------------------------------------------------------------------
__global__ __launch_bounds__(256) void weights_kernel(
    const float* __restrict__ Sc, const float* __restrict__ Qc,
    const int* __restrict__ cnt, float* __restrict__ w)
{
    const int d = blockIdx.x * 256 + threadIdx.x;
    if (d >= D_) return;
    float T = 0.f, Qtot = 0.f, posv = 0.f, sumC2 = 0.f;
    for (int c = 0; c < NC_; ++c) {
        float Scd = Sc[(size_t)c * D_ + d];
        float Qcd = Qc[(size_t)c * D_ + d];
        float cc  = (float)cnt[c];
        T     += Scd;
        Qtot  += Qcd;
        posv  += cc * Qcd - Scd * Scd;     // cnt_c*Q_c - S_c^2
        sumC2 += cc * cc;
    }
    posv *= 2.f;                                        // pos masked sqdiff
    float negv = 2.f * (float)N_ * Qtot - 2.f * T * T - posv;
    float cpos = sumC2 - (float)N_;
    float cneg = (float)N_ * (float)N_ - sumC2;
    float sig1 = (posv + EPS_) / cpos;
    float sig0 = (negv + EPS_) / cneg;
    w[d] = 1.f / sig0 - 1.f / sig1;
}

// ---------------------------------------------------------------------------
// K3: sq_i = sum_d w_d * x_id^2 ; one wave per row.
// ---------------------------------------------------------------------------
__global__ __launch_bounds__(256) void sq_kernel(
    const float* __restrict__ X, const float* __restrict__ w,
    float* __restrict__ sq)
{
    const int wvid = threadIdx.x >> 5;
    const int lane = threadIdx.x & 31;
    const int i    = blockIdx.x * 8 + wvid;
    const float* row = X + (size_t)i * D_;
    float a = 0.f;
    for (int d = lane; d < D_; d += 32) {
        float v = row[d];
        a += w[d] * v * v;
    }
    #pragma unroll
    for (int off = 16; off > 0; off >>= 1) a += __shfl_xor(a, off, 32);
    if (lane == 0) sq[i] = a;
}

// ---------------------------------------------------------------------------
// K4: one 256-thread block (8 waves) per upper-triangular 128x128 tile of S.
// G = (X*w) @ X^T via f16 hi/lo split: G ~= Ah.Bh + Ah.Bl + Al.Bh (f32 acc).
// Each wave owns a 64x32 tile (2x4 wave grid) = 2x4 WMMA C-tiles; the 4 A
// fragment hi/lo pairs are cached in registers and reused across both column
// tiles: 24 ds_load_b128 per 24 WMMAs (1.0/WMMA), ~195 VGPRs -> no spills.
// ---------------------------------------------------------------------------
__global__ __launch_bounds__(256) void gram_loss_kernel(
    const float* __restrict__ X, const int* __restrict__ tgt,
    const float* __restrict__ w, const float* __restrict__ sq,
    float* __restrict__ partial)
{
    __shared__ _Float16 Ah[TILE * LSTR];   // 10 KB each
    __shared__ _Float16 Al[TILE * LSTR];
    __shared__ _Float16 Bh[TILE * LSTR];
    __shared__ _Float16 Bl[TILE * LSTR];
    __shared__ float    wS[D_];            // 4 KB
    __shared__ float    sqI[TILE], sqJ[TILE];
    __shared__ int      tIs[TILE], tJs[TILE];
    __shared__ float    redS[8];

    const int tid = threadIdx.x;

    // decode linear block id -> (bi, bj), bi <= bj
    int q = blockIdx.x, bi = 0;
    while (q >= (NT - bi)) { q -= (NT - bi); ++bi; }
    const int bj = bi + q;
    const int i0 = bi * TILE, j0 = bj * TILE;

    for (int d = tid; d < D_; d += 256) wS[d] = w[d];
    if (tid < TILE) {
        sqI[tid] = sq[i0 + tid];  sqJ[tid] = sq[j0 + tid];
        tIs[tid] = tgt[i0 + tid]; tJs[tid] = tgt[j0 + tid];
    }
    __syncthreads();

    const int wvid  = tid >> 5;     // wave 0..7
    const int lane  = tid & 31;
    const int lhalf = lane >> 4;    // 0/1
    const int lm    = lane & 15;
    const int mrow  = (wvid & 1) * 64;    // 2x4 wave grid over 128x128
    const int ncol  = (wvid >> 1) * 32;

    v8f acc[8];                     // [nt*4 + mt], 64 VGPRs
    #pragma unroll
    for (int t = 0; t < 8; ++t)
        #pragma unroll
        for (int r = 0; r < 8; ++r) acc[t][r] = 0.f;

    // cooperative panel staging: 2 threads per row, 16 halves each
    const int lrow  = tid >> 1;     // 0..127
    const int lpart = tid & 1;      // halves [lpart*16, +16)
    const float* arow = X + (size_t)(i0 + lrow) * D_ + lpart * 16;
    const float* brow = X + (size_t)(j0 + lrow) * D_ + lpart * 16;
    _Float16* ahp = Ah + lrow * LSTR + lpart * 16;
    _Float16* alp = Al + lrow * LSTR + lpart * 16;
    _Float16* bhp = Bh + lrow * LSTR + lpart * 16;
    _Float16* blp = Bl + lrow * LSTR + lpart * 16;

    for (int kk = 0; kk < D_; kk += KT) {
        // ---- stage panels: fp32 load -> scale by w -> f16 hi/lo split ----
        const float4* ap = (const float4*)(arow + kk);
        const float4* bp = (const float4*)(brow + kk);
        float av[16], bv[16];
        #pragma unroll
        for (int v = 0; v < 4; ++v) {
            float4 a4 = ap[v], b4 = bp[v];
            av[4*v+0] = a4.x; av[4*v+1] = a4.y; av[4*v+2] = a4.z; av[4*v+3] = a4.w;
            bv[4*v+0] = b4.x; bv[4*v+1] = b4.y; bv[4*v+2] = b4.z; bv[4*v+3] = b4.w;
        }
        if (kk + KT < D_) {                       // global_prefetch_b8 next panel
            __builtin_prefetch(arow + kk + KT, 0, 1);
            __builtin_prefetch(brow + kk + KT, 0, 1);
        }
        #pragma unroll
        for (int e = 0; e < 16; ++e) {
            float ww = wS[kk + lpart * 16 + e];
            float xw = av[e] * ww;
            _Float16 hf = (_Float16)xw;
            ahp[e] = hf;
            alp[e] = (_Float16)(xw - (float)hf);
            float xb = bv[e];
            _Float16 hb = (_Float16)xb;
            bhp[e] = hb;
            blp[e] = (_Float16)(xb - (float)hb);
        }
        __syncthreads();

        // ---- cache this wave's 4 A fragment hi/lo pairs in registers ----
        // 16-bit A 16x32 layout (ISA 7.12.2): element e -> K =
        // (e<8 ? e : e+8) + 8*half
        v16h afh[4], afl[4];
        #pragma unroll
        for (int mt = 0; mt < 4; ++mt) {
            const _Float16* abh = Ah + (mrow + mt * 16 + lm) * LSTR;
            const _Float16* abl = Al + (mrow + mt * 16 + lm) * LSTR;
            #pragma unroll
            for (int e = 0; e < 16; ++e) {
                int k = (e < 8 ? e : e + 8) + 8 * lhalf;
                afh[mt][e] = abh[k];
                afl[mt][e] = abl[k];
            }
        }

        // ---- 2 column tiles x 4 row tiles x 3 split WMMAs ----
        #pragma unroll
        for (int nt = 0; nt < 2; ++nt) {
            // 16-bit B layout (matches A): element e -> K = (e<8 ? e : e+8) + 8*half
            v16h bfh, bfl;
            const _Float16* bbh = Bh + (ncol + nt * 16 + lm) * LSTR;
            const _Float16* bbl = Bl + (ncol + nt * 16 + lm) * LSTR;
            #pragma unroll
            for (int e = 0; e < 16; ++e) {
                int k = (e < 8 ? e : e + 8) + 8 * lhalf;
                bfh[e] = bbh[k];
                bfl[e] = bbl[k];
            }
            #pragma unroll
            for (int mt = 0; mt < 4; ++mt) {
                const int t = nt * 4 + mt;
                acc[t] = __builtin_amdgcn_wmma_f32_16x16x32_f16(
                    false, afh[mt], false, bfh, (short)0, acc[t], false, false);
                acc[t] = __builtin_amdgcn_wmma_f32_16x16x32_f16(
                    false, afh[mt], false, bfl, (short)0, acc[t], false, false);
                acc[t] = __builtin_amdgcn_wmma_f32_16x16x32_f16(
                    false, afl[mt], false, bfh, (short)0, acc[t], false, false);
            }
        }
        __syncthreads();
    }

    // ---- epilogue: S = sq_i + sq_j - 2G, masked softplus ----
    float lsum = 0.f;
    #pragma unroll
    for (int nt = 0; nt < 2; ++nt) {
        const int   jn = ncol + nt * 16 + lm;
        const int   tj = tJs[jn];
        const float sj = sqJ[jn];
        const int   gj = j0 + jn;
        #pragma unroll
        for (int mt = 0; mt < 4; ++mt) {
            #pragma unroll
            for (int r = 0; r < 8; ++r) {
                // C layout: VGPR r, lanes 0-15 -> M=r, lanes 16-31 -> M=r+8
                int   il = mrow + mt * 16 + r + 8 * lhalf;
                float s  = sqI[il] + sj - 2.f * acc[nt * 4 + mt][r];
                float c;
                if (tIs[il] == tj)
                    c = ((i0 + il) == gj) ? 0.f : softplus_f(-s);  // pos pair
                else
                    c = softplus_f(s);                             // neg pair
                lsum += c;
            }
        }
    }
    if (bi != bj) lsum *= 2.f;   // off-diagonal tile counted for (i,j)+(j,i)

    #pragma unroll
    for (int off = 16; off > 0; off >>= 1) lsum += __shfl_xor(lsum, off, 32);
    if (lane == 0) redS[wvid] = lsum;
    __syncthreads();
    if (tid == 0) {
        float t = 0.f;
        for (int v = 0; v < 8; ++v) t += redS[v];
        partial[blockIdx.x] = t;
    }
}

// ---------------------------------------------------------------------------
// K5: fixed-order final reduction; denominator = cnt_pos+cnt_neg = N^2 - N.
// ---------------------------------------------------------------------------
__global__ __launch_bounds__(256) void final_kernel(
    const float* __restrict__ partial, float* __restrict__ out)
{
    __shared__ float red[256];
    const int tid = threadIdx.x;
    float a = 0.f;
    for (int p = tid; p < NTILES; p += 256) a += partial[p];
    red[tid] = a;
    __syncthreads();
    for (int s = 128; s > 0; s >>= 1) {
        if (tid < s) red[tid] += red[tid + s];
        __syncthreads();
    }
    if (tid == 0) {
        float denom = (float)N_ * (float)N_ - (float)N_;
        out[0] = red[0] / denom;
    }
}

// ---------------------------------------------------------------------------
extern "C" void kernel_launch(void* const* d_in, const int* in_sizes, int n_in,
                              void* d_out, int out_size, void* d_ws, size_t ws_size,
                              hipStream_t stream) {
    (void)in_sizes; (void)n_in; (void)out_size; (void)ws_size;
    const float* X   = (const float*)d_in[0];
    const int*   tgt = (const int*)d_in[1];
    float*       out = (float*)d_out;

    // workspace layout (~548 KB)
    float* ws      = (float*)d_ws;
    float* Sc      = ws;                    // 64*1024
    float* Qc      = Sc + (size_t)NC_ * D_; // 64*1024
    float* w       = Qc + (size_t)NC_ * D_; // 1024
    float* sq      = w + D_;                // 4096
    float* partial = sq + N_;               // 528
    int*   cnt     = (int*)(partial + NTILES); // 64 ints

    class_stats_kernel<<<NC_, 256, 0, stream>>>(X, tgt, Sc, Qc, cnt);
    weights_kernel<<<D_ / 256, 256, 0, stream>>>(Sc, Qc, cnt, w);
    sq_kernel<<<N_ / 8, 256, 0, stream>>>(X, w, sq);
    gram_loss_kernel<<<NTILES, 256, 0, stream>>>(X, tgt, w, sq, partial);
    final_kernel<<<1, 256, 0, stream>>>(partial, out);
}
